// GATheadClassifier_55027120997065
// MI455X (gfx1250) — matrix operand order repaired and checked by reading, not compiled, not run on hardware
//
#include <hip/hip_runtime.h>
#include <math.h>

// ---------------- problem constants (from reference setup_inputs) -----------
#define BQ   16
#define NQ   1024
#define HQ   256
#define H2Q  512
#define LQ   8
#define NTQ  (BQ * NQ)              // 16384 total nodes
#define NE1Q (BQ * (NQ - 1))        // 16368 MST edges (one direction)
#define NEQ  (2 * NE1Q + NTQ)       // 49120 edges incl. reverse + self loops
#define INFQ 1e30f
#define ALPHAQ 0.3f

typedef __attribute__((ext_vector_type(2)))  float          v2f;
typedef __attribute__((ext_vector_type(8)))  float          v8f;
typedef __attribute__((ext_vector_type(16))) __bf16         v16bf;
typedef __attribute__((ext_vector_type(16))) unsigned short v16u;

// ---------------- helpers ---------------------------------------------------
__device__ inline void atomicAddF(float* p, float v) {
    __hip_atomic_fetch_add(p, v, __ATOMIC_RELAXED, __HIP_MEMORY_SCOPE_AGENT);
}

__device__ inline unsigned short f2bf(float f) {   // round-to-nearest-even
    unsigned u = __builtin_bit_cast(unsigned, f);
    unsigned r = u + 0x7fffu + ((u >> 16) & 1u);
    return (unsigned short)(r >> 16);
}

__device__ inline void wave_argmin(float& v, int& i) {
    #pragma unroll
    for (int off = 16; off > 0; off >>= 1) {
        float ov = __shfl_down(v, off, 32);
        int   oi = __shfl_down(i, off, 32);
        if (ov < v || (ov == v && oi < i)) { v = ov; i = oi; }
    }
}

// ---------------- small utility kernels -------------------------------------
__global__ void zero_kernel(float* p, int n) {
    int i = blockIdx.x * blockDim.x + threadIdx.x;
    if (i < n) p[i] = 0.f;
}

__global__ void cast_bf16_kernel(const float* __restrict__ s, unsigned short* __restrict__ d, int n) {
    int i = blockIdx.x * blockDim.x + threadIdx.x;
    if (i < n) d[i] = f2bf(s[i]);
}

// n2[node] = sum_k x[node][k]^2 ; one wave per node
__global__ void norms_kernel(const float* __restrict__ X, float* __restrict__ n2) {
    int node = blockIdx.x * 8 + (threadIdx.x >> 5);
    int lane = threadIdx.x & 31;
    const float* p = X + (size_t)node * HQ;
    float s = 0.f;
    #pragma unroll
    for (int t = 0; t < HQ; t += 32) { float v = p[t + lane]; s += v * v; }
    #pragma unroll
    for (int off = 16; off > 0; off >>= 1) s += __shfl_down(s, off, 32);
    if (lane == 0) n2[node] = s;
}

// ---------------- cdist via V_WMMA_F32_16X16X4_F32 --------------------------
// One wave computes a 16x16 tile of dist[b]. Gram G = X X^T, then
// dist = sqrt(max(n2_i + n2_j - 2G, 0)).  A(16x4 f32): lanes0-15 hold K=0,1
// in the 2 VGPRs, lanes16-31 hold K=2,3.  B(4x16) mirrors with N=lane&15.
__global__ void cdist_wmma(const float* __restrict__ X, const float* __restrict__ n2,
                           float* __restrict__ dist) {
    int b = blockIdx.z;
    int i0 = blockIdx.y * 16, j0 = blockIdx.x * 16;
    int lane = threadIdx.x;
    int half = lane >> 4, lm = lane & 15;
    const float* Xb = X + (size_t)b * NQ * HQ;
    v8f c = {};
    for (int k0 = 0; k0 < HQ; k0 += 4) {
        int k = k0 + half * 2;
        const float* pa = Xb + (size_t)(i0 + lm) * HQ + k;
        const float* pb = Xb + (size_t)(j0 + lm) * HQ + k;
        v2f a = { pa[0], pa[1] };
        v2f bb = { pb[0], pb[1] };
        c = __builtin_amdgcn_wmma_f32_16x16x4_f32(false, a, false, bb,
                                                  (short)0, c, false, false);
    }
    #pragma unroll
    for (int r = 0; r < 8; ++r) {
        int i = i0 + r + half * 8;
        int j = j0 + lm;
        float d2 = n2[b * NQ + i] + n2[b * NQ + j] - 2.f * c[r];
        dist[((size_t)b * NQ + i) * NQ + j] = sqrtf(fmaxf(d2, 0.f));
    }
}

// ---------------- dense Prim MST: one block per batch -----------------------
__global__ __launch_bounds__(1024)
void prim_kernel(const float* __restrict__ dist, int* __restrict__ ps,
                 int* __restrict__ vs, float* __restrict__ wts) {
    int b = blockIdx.x, tid = threadIdx.x;
    const float* Db = dist + (size_t)b * NQ * NQ;
    float mind = Db[tid];           // dist[0][tid]
    int   par  = 0;
    bool  inTree = (tid == 0);
    int wid = tid >> 5, lane = tid & 31;
    __shared__ float sval[32];
    __shared__ int   sidx[32];
    __shared__ float s_w;
    __shared__ int   s_v, s_p;
    for (int t = 0; t < NQ - 1; ++t) {
        float v = inTree ? INFQ : mind;
        int idx = tid;
        wave_argmin(v, idx);
        if (lane == 0) { sval[wid] = v; sidx[wid] = idx; }
        __syncthreads();
        if (wid == 0) {
            float v2 = sval[lane];
            int   i2 = sidx[lane];
            wave_argmin(v2, i2);
            if (lane == 0) { s_w = v2; s_v = i2; }
        }
        __syncthreads();
        int vsel = s_v;
        if (tid == vsel) { s_p = par; inTree = true; }
        __syncthreads();
        if (tid == 0) {
            ps[b * (NQ - 1) + t]  = s_p;
            vs[b * (NQ - 1) + t]  = vsel;
            wts[b * (NQ - 1) + t] = s_w;
        }
        float nd = Db[(size_t)vsel * NQ + tid];
        if (nd < mind) { mind = nd; par = vsel; }
    }
}

// ---------------- edge list (+reverse +self loops) & degree ------------------
__global__ void edges_kernel(const int* __restrict__ ps, const int* __restrict__ vs,
                             const float* __restrict__ wts, int* __restrict__ esrc,
                             int* __restrict__ edst, float* __restrict__ ewt,
                             float* __restrict__ deg) {
    int idx = blockIdx.x * blockDim.x + threadIdx.x;
    if (idx < NE1Q) {
        int b = idx / (NQ - 1), off = b * NQ;
        int p = ps[idx] + off, v = vs[idx] + off;
        float w = wts[idx];
        esrc[idx] = p;  edst[idx] = v;  ewt[idx] = w;
        esrc[NE1Q + idx] = v;  edst[NE1Q + idx] = p;  ewt[NE1Q + idx] = w;
        atomicAddF(deg + v, w);
        atomicAddF(deg + p, w);
    }
    if (idx < NTQ) {
        int e = 2 * NE1Q + idx;
        esrc[e] = idx;  edst[e] = idx;  ewt[e] = 1.f;
        atomicAddF(deg + idx, 1.f);
    }
}

__global__ void dinv_kernel(const float* __restrict__ deg, float* __restrict__ dinv) {
    int i = blockIdx.x * blockDim.x + threadIdx.x;
    if (i < NTQ) {
        float d = deg[i];
        dinv[i] = (d > 0.f) ? rsqrtf(fmaxf(d, 1e-30f)) : 0.f;
    }
}

__global__ void enorm_kernel(const int* __restrict__ esrc, const int* __restrict__ edst,
                             const float* __restrict__ ewt, const float* __restrict__ dinv,
                             float* __restrict__ enorm) {
    int e = blockIdx.x * blockDim.x + threadIdx.x;
    if (e < NEQ) enorm[e] = dinv[esrc[e]] * ewt[e] * dinv[edst[e]];
}

// ---------------- sparse propagation: prop[dst] += norm * x[src] -------------
__global__ void prop_scatter(const float* __restrict__ x, const int* __restrict__ esrc,
                             const int* __restrict__ edst, const float* __restrict__ enorm,
                             float* __restrict__ prop, int F) {
    int chunks = F >> 2;
    int idx = blockIdx.x * blockDim.x + threadIdx.x;
    if (idx >= NEQ * chunks) return;
    int e = idx / chunks;
    int c = (idx - e * chunks) << 2;
    int s = esrc[e], d = edst[e];
    float nm = enorm[e];
    const float4 xv = *(const float4*)(x + (size_t)s * F + c);
    float* pd = prop + (size_t)d * F + c;
    atomicAddF(pd + 0, nm * xv.x);
    atomicAddF(pd + 1, nm * xv.y);
    atomicAddF(pd + 2, nm * xv.z);
    atomicAddF(pd + 3, nm * xv.w);
}

// h = alpha*x + (1-alpha)*prop  -> bf16
__global__ void hform_kernel(const float* __restrict__ x, const float* __restrict__ prop,
                             unsigned short* __restrict__ h, int n) {
    int i = blockIdx.x * blockDim.x + threadIdx.x;
    if (i < n) h[i] = f2bf(ALPHAQ * x[i] + (1.f - ALPHAQ) * prop[i]);
}

// ---------------- GEMM: out = tanh(A_bf16 @ W_bf16 + bias), f32 accum --------
// One wave per 16x16 output tile. bf16 A(16x32) lane layout per ISA §7.12.2:
// lane<16: elems 0..7 -> K 0..7, elems 8..15 -> K 16..23; lane>=16 shifts K by 8.
// bf16 B(32x16): same K layout as A (col = lane&15).
__global__ void gemm_bf16_tanh(const unsigned short* __restrict__ A,
                               const unsigned short* __restrict__ W,
                               const float* __restrict__ bias,
                               float* __restrict__ out, int K, int Nc) {
    int lane = threadIdx.x;
    int half = lane >> 4, lm = lane & 15;
    int j0 = blockIdx.x * 16, i0 = blockIdx.y * 16;
    int m = i0 + lm, col = j0 + lm;
    v8f c = {};
    for (int k0 = 0; k0 < K; k0 += 32) {
        v16u au, bu;
        #pragma unroll
        for (int e = 0; e < 16; ++e) {
            int ka = k0 + e + ((e >= 8) ? 8 : 0) + (half ? 8 : 0);
            int kb = k0 + e + ((e >= 8) ? 8 : 0) + (half ? 8 : 0);
            au[e] = A[(size_t)m * K + ka];
            bu[e] = W[(size_t)kb * Nc + col];
        }
        v16bf a  = __builtin_bit_cast(v16bf, au);
        v16bf bb = __builtin_bit_cast(v16bf, bu);
        c = __builtin_amdgcn_wmma_f32_16x16x32_bf16(false, a, false, bb,
                                                    (short)0, c, false, false);
    }
    #pragma unroll
    for (int r = 0; r < 8; ++r) {
        int row = i0 + r + half * 8;
        out[(size_t)row * Nc + col] = tanhf(c[r] + bias[col]);
    }
}

// ---------------- pooling + MLP head -----------------------------------------
__global__ void pool_kernel(const float* __restrict__ x, float* __restrict__ pooled) {
    int b = blockIdx.x, f = threadIdx.x;      // 512 threads
    float s = 0.f;
    for (int n = 0; n < NQ; ++n) s += x[((size_t)b * NQ + n) * H2Q + f];
    pooled[b * H2Q + f] = s * (1.0f / NQ);
}

__global__ void head_kernel(const float* __restrict__ pooled, const float* __restrict__ Wd,
                            const float* __restrict__ bd, const float* __restrict__ Wo,
                            const float* __restrict__ bo, float* __restrict__ out) {
    __shared__ float hd[HQ];
    int b = blockIdx.x, t = threadIdx.x;      // 256 threads
    float acc = bd[t];
    const float* pr = pooled + b * H2Q;
    for (int k = 0; k < H2Q; ++k) acc += pr[k] * Wd[k * HQ + t];
    hd[t] = tanhf(acc);
    __syncthreads();
    if (t < LQ) {
        float o = bo[t];
        for (int k = 0; k < HQ; ++k) o += hd[k] * Wo[k * LQ + t];
        out[b * LQ + t] = o;
    }
}

// ---------------- workspace layout (bytes) -----------------------------------
// dist (64MB, dead after Prim) overlaps xa(32MB)+prop(32MB) used afterwards.
static constexpr size_t OFF_XA   = 0;                               // NT*H2 f32
static constexpr size_t OFF_PROP = (size_t)NTQ * H2Q * 4;           // NT*H2 f32
static constexpr size_t OFF_DIST = 0;                               // B*N*N f32
static constexpr size_t OFF_XB   = (size_t)BQ * NQ * NQ * 4;        // NT*H2 f32
static constexpr size_t OFF_HBF  = OFF_XB + (size_t)NTQ * H2Q * 4;  // NT*H2 bf16
static constexpr size_t OFF_N2   = OFF_HBF + (size_t)NTQ * H2Q * 2;
static constexpr size_t OFF_PS   = OFF_N2   + (size_t)NTQ * 4;
static constexpr size_t OFF_VS   = OFF_PS   + (size_t)NE1Q * 4;
static constexpr size_t OFF_WTS  = OFF_VS   + (size_t)NE1Q * 4;
static constexpr size_t OFF_ESRC = OFF_WTS  + (size_t)NE1Q * 4;
static constexpr size_t OFF_EDST = OFF_ESRC + (size_t)NEQ * 4;
static constexpr size_t OFF_EWT  = OFF_EDST + (size_t)NEQ * 4;
static constexpr size_t OFF_ENRM = OFF_EWT  + (size_t)NEQ * 4;
static constexpr size_t OFF_DEG  = OFF_ENRM + (size_t)NEQ * 4;
static constexpr size_t OFF_DINV = OFF_DEG  + (size_t)NTQ * 4;
static constexpr size_t OFF_W1B  = OFF_DINV + (size_t)NTQ * 4;
static constexpr size_t OFF_W2B  = OFF_W1B  + (size_t)HQ * H2Q * 2;
static constexpr size_t OFF_W3B  = OFF_W2B  + (size_t)H2Q * H2Q * 2;
static constexpr size_t OFF_POOL = OFF_W3B  + (size_t)H2Q * H2Q * 2;

extern "C" void kernel_launch(void* const* d_in, const int* in_sizes, int n_in,
                              void* d_out, int out_size, void* d_ws, size_t ws_size,
                              hipStream_t stream) {
    (void)in_sizes; (void)n_in; (void)out_size; (void)ws_size;
    const float* feat = (const float*)d_in[0];
    const float* W1 = (const float*)d_in[1];  const float* b1 = (const float*)d_in[2];
    const float* W2 = (const float*)d_in[3];  const float* b2 = (const float*)d_in[4];
    const float* W3 = (const float*)d_in[5];  const float* b3 = (const float*)d_in[6];
    const float* Wd = (const float*)d_in[7];  const float* bd = (const float*)d_in[8];
    const float* Wo = (const float*)d_in[9];  const float* bo = (const float*)d_in[10];

    char* ws = (char*)d_ws;
    float*          xa    = (float*)(ws + OFF_XA);
    float*          prop  = (float*)(ws + OFF_PROP);
    float*          dist  = (float*)(ws + OFF_DIST);
    float*          xb    = (float*)(ws + OFF_XB);
    unsigned short* hbf   = (unsigned short*)(ws + OFF_HBF);
    float*          n2    = (float*)(ws + OFF_N2);
    int*            ps    = (int*)(ws + OFF_PS);
    int*            vs    = (int*)(ws + OFF_VS);
    float*          wts   = (float*)(ws + OFF_WTS);
    int*            esrc  = (int*)(ws + OFF_ESRC);
    int*            edst  = (int*)(ws + OFF_EDST);
    float*          ewt   = (float*)(ws + OFF_EWT);
    float*          enrm  = (float*)(ws + OFF_ENRM);
    float*          deg   = (float*)(ws + OFF_DEG);
    float*          dinv  = (float*)(ws + OFF_DINV);
    unsigned short* W1b   = (unsigned short*)(ws + OFF_W1B);
    unsigned short* W2b   = (unsigned short*)(ws + OFF_W2B);
    unsigned short* W3b   = (unsigned short*)(ws + OFF_W3B);
    float*          pooled= (float*)(ws + OFF_POOL);

    const int TB = 256;
    // ---- phase 0: norms, weight casts, degree init --------------------------
    zero_kernel<<<(NTQ + TB - 1) / TB, TB, 0, stream>>>(deg, NTQ);
    norms_kernel<<<NTQ / 8, TB, 0, stream>>>(feat, n2);
    cast_bf16_kernel<<<(HQ * H2Q + TB - 1) / TB, TB, 0, stream>>>(W1, W1b, HQ * H2Q);
    cast_bf16_kernel<<<(H2Q * H2Q + TB - 1) / TB, TB, 0, stream>>>(W2, W2b, H2Q * H2Q);
    cast_bf16_kernel<<<(H2Q * H2Q + TB - 1) / TB, TB, 0, stream>>>(W3, W3b, H2Q * H2Q);

    // ---- phase 1: distance matrix (f32 WMMA) + Prim MST ---------------------
    cdist_wmma<<<dim3(NQ / 16, NQ / 16, BQ), 32, 0, stream>>>(feat, n2, dist);
    prim_kernel<<<BQ, 1024, 0, stream>>>(dist, ps, vs, wts);

    // ---- phase 2: edges + gcn_norm ------------------------------------------
    edges_kernel<<<(NTQ + TB - 1) / TB, TB, 0, stream>>>(ps, vs, wts, esrc, edst, ewt, deg);
    dinv_kernel<<<(NTQ + TB - 1) / TB, TB, 0, stream>>>(deg, dinv);
    enorm_kernel<<<(NEQ + TB - 1) / TB, TB, 0, stream>>>(esrc, edst, ewt, dinv, enrm);

    // ---- layer 1: H -> H2 (dist region is dead; xa/prop alias it) -----------
    zero_kernel<<<(NTQ * HQ + TB - 1) / TB, TB, 0, stream>>>(prop, NTQ * HQ);
    prop_scatter<<<(NEQ * (HQ / 4) + TB - 1) / TB, TB, 0, stream>>>(feat, esrc, edst, enrm, prop, HQ);
    hform_kernel<<<(NTQ * HQ + TB - 1) / TB, TB, 0, stream>>>(feat, prop, hbf, NTQ * HQ);
    gemm_bf16_tanh<<<dim3(H2Q / 16, NTQ / 16), 32, 0, stream>>>(hbf, W1b, b1, xa, HQ, H2Q);

    // ---- layer 2: H2 -> H2 ---------------------------------------------------
    zero_kernel<<<(NTQ * H2Q + TB - 1) / TB, TB, 0, stream>>>(prop, NTQ * H2Q);
    prop_scatter<<<(NEQ * (H2Q / 4) + TB - 1) / TB, TB, 0, stream>>>(xa, esrc, edst, enrm, prop, H2Q);
    hform_kernel<<<(NTQ * H2Q + TB - 1) / TB, TB, 0, stream>>>(xa, prop, hbf, NTQ * H2Q);
    gemm_bf16_tanh<<<dim3(H2Q / 16, NTQ / 16), 32, 0, stream>>>(hbf, W2b, b2, xb, H2Q, H2Q);

    // ---- layer 3: H2 -> H2 ---------------------------------------------------
    zero_kernel<<<(NTQ * H2Q + TB - 1) / TB, TB, 0, stream>>>(prop, NTQ * H2Q);
    prop_scatter<<<(NEQ * (H2Q / 4) + TB - 1) / TB, TB, 0, stream>>>(xb, esrc, edst, enrm, prop, H2Q);
    hform_kernel<<<(NTQ * H2Q + TB - 1) / TB, TB, 0, stream>>>(xb, prop, hbf, NTQ * H2Q);
    gemm_bf16_tanh<<<dim3(H2Q / 16, NTQ / 16), 32, 0, stream>>>(hbf, W3b, b3, xa, H2Q, H2Q);

    // ---- pool + head ---------------------------------------------------------
    pool_kernel<<<BQ, H2Q, 0, stream>>>(xa, pooled);
    head_kernel<<<BQ, HQ, 0, stream>>>(pooled, Wd, bd, Wo, bo, (float*)d_out);
}
